// CausalSelfAttentionMLA_37185826849440
// MI455X (gfx1250) — matrix operand order repaired and checked by reading, not compiled, not run on hardware
//
#include <hip/hip_runtime.h>
#include <hip/hip_bf16.h>

// ---------------- problem constants ----------------
#define B_DIM   2
#define L_DIM   2048
#define HID_DIM 2048
#define NH_DIM  16
#define NKV_DIM 4
#define HD_DIM  128
#define LAT_DIM 512
#define QPK     (NH_DIM / NKV_DIM)          // 4
#define ATT_SCALE 0.088388347648318447f     // 128^-0.5
#define ROPE_THETA 10000.0f

typedef __attribute__((ext_vector_type(16))) __bf16 v16bf;
typedef __attribute__((ext_vector_type(8)))  float  v8f;
typedef __attribute__((ext_vector_type(4)))  int    v4i;

union Frag16 { v16bf v; uint4 u[2]; };

// ---------------- gfx1250 async global->LDS copy ----------------
typedef __attribute__((address_space(1))) v4i as1_v4i;
typedef __attribute__((address_space(3))) v4i as3_v4i;

#if __has_builtin(__builtin_amdgcn_global_load_async_to_lds_b128)
#define HAVE_ASYNC_LDS 1
#endif

#if defined(HAVE_ASYNC_LDS)
#if __has_builtin(__builtin_amdgcn_s_wait_asynccnt)
#define ASYNC_WAIT(n) __builtin_amdgcn_s_wait_asynccnt(n)
#else
#define ASYNC_WAIT(n) asm volatile("s_wait_asynccnt %0" ::"i"(n) : "memory")
#endif
#else
#define ASYNC_WAIT(n)
#endif

static __device__ __forceinline__ void async_copy16(const void* g, void* l) {
#if defined(HAVE_ASYNC_LDS)
  __builtin_amdgcn_global_load_async_to_lds_b128((as1_v4i*)g, (as3_v4i*)l, 0, 0);
#else
  *reinterpret_cast<uint4*>(l) = *reinterpret_cast<const uint4*>(g);
#endif
}

static __device__ __forceinline__ unsigned short f2bf(float f) {
  unsigned int u = __float_as_uint(f);
  u += 0x7FFFu + ((u >> 16) & 1u);          // round-to-nearest-even
  return (unsigned short)(u >> 16);
}

static __device__ __forceinline__ v8f zero8() {
  v8f z;
#pragma unroll
  for (int i = 0; i < 8; ++i) z[i] = 0.0f;
  return z;
}

// ---------------- f32 -> bf16 conversion (4 elems/thread) ----------------
__global__ void cvt_f32_bf16_x4(const float* __restrict__ in,
                                unsigned short* __restrict__ out,
                                unsigned long long n4) {
  unsigned long long i = (unsigned long long)blockIdx.x * blockDim.x + threadIdx.x;
  if (i >= n4) return;
  float4 f = reinterpret_cast<const float4*>(in)[i];
  union { unsigned short s[4]; uint2 u; } r;
  r.s[0] = f2bf(f.x); r.s[1] = f2bf(f.y); r.s[2] = f2bf(f.z); r.s[3] = f2bf(f.w);
  reinterpret_cast<uint2*>(out)[i] = r.u;
}

// ---------------- f32 [K][N] -> bf16 [N][K] weight transpose ----------------
__global__ __launch_bounds__(256)
void transpose_f32_bf16(const float* __restrict__ in,
                        unsigned short* __restrict__ out, int K, int N) {
  __shared__ float t[32][33];
  const int k0 = blockIdx.y * 32, n0 = blockIdx.x * 32;
  const int tx = threadIdx.x & 31, ty = threadIdx.x >> 5;   // ty: 0..7
#pragma unroll
  for (int i = ty; i < 32; i += 8)
    t[i][tx] = in[(size_t)(k0 + i) * N + n0 + tx];
  __syncthreads();
#pragma unroll
  for (int i = ty; i < 32; i += 8)
    out[(size_t)(n0 + i) * K + k0 + tx] = f2bf(t[tx][i]);
}

// --------- V transpose: f32 [B,L,NKV,HD] -> bf16 [B,NKV,HD,L] ---------
__global__ __launch_bounds__(256)
void transpose_v_bf16(const float* __restrict__ in,
                      unsigned short* __restrict__ out) {
  __shared__ float t[32][33];
  const int bk = blockIdx.z;                 // b*NKV + kvh
  const int b = bk / NKV_DIM, kvh = bk % NKV_DIM;
  const int l0 = blockIdx.y * 32, d0 = blockIdx.x * 32;
  const int tx = threadIdx.x & 31, ty = threadIdx.x >> 5;
#pragma unroll
  for (int i = ty; i < 32; i += 8)
    t[i][tx] = in[(((size_t)b * L_DIM + l0 + i) * NKV_DIM + kvh) * HD_DIM + d0 + tx];
  __syncthreads();
#pragma unroll
  for (int i = ty; i < 32; i += 8)
    out[(((size_t)b * NKV_DIM + kvh) * HD_DIM + d0 + i) * L_DIM + l0 + tx] =
        f2bf(t[tx][i]);
}

// ---------------- RoPE (+ f32 -> bf16) ----------------
__global__ void rope_to_bf16(const float* __restrict__ in,
                             unsigned short* __restrict__ out, int nh) {
  unsigned long long idx = (unsigned long long)blockIdx.x * blockDim.x + threadIdx.x;
  unsigned long long total = (unsigned long long)B_DIM * L_DIM * nh * 64;
  if (idx >= total) return;
  int d = (int)(idx & 63u);
  unsigned long long t = idx >> 6;          // (b*L + pos)*nh + h
  unsigned long long row = t / nh;          // b*L + pos
  int pos = (int)(row % L_DIM);
  unsigned long long base = t * HD_DIM;
  float invf = __powf(ROPE_THETA, -(float)d * (1.0f / 64.0f));
  float fr = (float)pos * invf;
  float s, c;
  __sincosf(fr, &s, &c);
  float t1 = in[base + d];
  float t2 = in[base + 64 + d];
  out[base + d]      = f2bf(t1 * c - t2 * s);
  out[base + 64 + d] = f2bf(t2 * c + t1 * s);
}

// ---------------- bf16 GEMM: C(f32)[M,N] = A[M,K] @ Bt[N,K]^T ----------------
// Block tile 128x128x32, 256 threads = 8 waves, wave tile 32x64 (2x4 WMMAs).
// Double-buffered LDS, async global->LDS staging.
#define BM 128
#define BN 128
#define BK 32

__global__ __launch_bounds__(256)
void gemm_bf16_wmma(const unsigned short* __restrict__ A,   // [M][K]
                    const unsigned short* __restrict__ Bt,  // [N][K]
                    float* __restrict__ C, int M, int N, int K) {
  __shared__ unsigned short As[2][BM][BK];  // 2 x 8 KB
  __shared__ unsigned short Bs[2][BN][BK];  // 2 x 8 KB
  const int tid   = threadIdx.x;
  const int lane  = tid & 31;
  const int wave  = tid >> 5;               // 0..7
  const int wm    = wave >> 1;              // 0..3 -> 32-row group
  const int wn    = wave & 1;               // 0..1 -> 64-col group
  const int l16   = lane & 15;
  const int lhalf = lane >> 4;
  const int bn0   = blockIdx.x * BN;
  const int bm0   = blockIdx.y * BM;

  v8f acc[2][4];
#pragma unroll
  for (int i = 0; i < 2; ++i)
#pragma unroll
    for (int j = 0; j < 4; ++j) acc[i][j] = zero8();

  const int srow = tid >> 1;                 // 0..127
  const int skoff = (tid & 1) * 16;          // 0 or 16
  auto stage = [&](int k0, int buf) {        // 4 async b128 per thread
    const unsigned short* ag = &A[(size_t)(bm0 + srow) * K + k0 + skoff];
    unsigned short* al = &As[buf][srow][skoff];
    async_copy16(ag, al);
    async_copy16(ag + 8, al + 8);
    const unsigned short* bg = &Bt[(size_t)(bn0 + srow) * K + k0 + skoff];
    unsigned short* bl = &Bs[buf][srow][skoff];
    async_copy16(bg, bl);
    async_copy16(bg + 8, bl + 8);
  };

  const int nk = K / BK;
  stage(0, 0);
  for (int t = 0; t < nk; ++t) {
    const int buf = t & 1;
    __syncthreads();                         // buf^1 readers done before overwrite
    if (t + 1 < nk) {
      stage((t + 1) * BK, buf ^ 1);
      ASYNC_WAIT(4);                         // in-order: tile t complete
    } else {
      ASYNC_WAIT(0);
    }
    __syncthreads();

    Frag16 afrag[2];
#pragma unroll
    for (int i = 0; i < 2; ++i) {
      int row = wm * 32 + i * 16 + l16;
      int kh  = lhalf * 8;                   // A 16x32: lanes 16-31 start at K=8
      afrag[i].u[0] = *reinterpret_cast<const uint4*>(&As[buf][row][kh]);
      afrag[i].u[1] = *reinterpret_cast<const uint4*>(&As[buf][row][kh + 16]);
    }
#pragma unroll
    for (int j = 0; j < 4; ++j) {
      Frag16 bfrag;
      int col = wn * 64 + j * 16 + l16;
      int ks  = lhalf * 8;                   // B 32x16: lanes 16-31 start at K=8
      bfrag.u[0] = *reinterpret_cast<const uint4*>(&Bs[buf][col][ks]);
      bfrag.u[1] = *reinterpret_cast<const uint4*>(&Bs[buf][col][ks + 16]);
#pragma unroll
      for (int i = 0; i < 2; ++i)
        acc[i][j] = __builtin_amdgcn_wmma_f32_16x16x32_bf16(
            false, afrag[i].v, false, bfrag.v, (short)0, acc[i][j], false, false);
    }
  }

#pragma unroll
  for (int i = 0; i < 2; ++i)
#pragma unroll
    for (int j = 0; j < 4; ++j)
#pragma unroll
      for (int v = 0; v < 8; ++v) {
        int row = bm0 + wm * 32 + i * 16 + lhalf * 8 + v;
        int col = bn0 + wn * 64 + j * 16 + l16;
        C[(size_t)row * N + col] = acc[i][j][v];
      }
}

// ---------------- flash attention (online softmax, WMMA) ----------------
// Q bf16 [B,L,NH,HD], K bf16 [B,L,NKV,HD], Vt bf16 [B,NKV,HD,L]; Y f32 [B,L,NH,HD].
// Block: 64 q-rows (4 waves x 16 rows), 32-key KV tiles, double-buffered async LDS.
__global__ __launch_bounds__(128)
void flash_attn_wmma(const unsigned short* __restrict__ Qb,
                     const unsigned short* __restrict__ Kb,
                     const unsigned short* __restrict__ Vtg,
                     const int* __restrict__ amask,
                     float* __restrict__ Y) {
  __shared__ unsigned short Ks[2][32][HD_DIM];  // 2 x 8 KB, (key, d)
  __shared__ unsigned short Vt[2][HD_DIM][32];  // 2 x 8 KB, (d, key)
  __shared__ unsigned short Ps[64][32];         // 4 KB
  const int tid   = threadIdx.x;
  const int lane  = tid & 31;
  const int wave  = tid >> 5;                   // 0..3
  const int l16   = lane & 15;
  const int lhalf = lane >> 4;
  const int bh  = blockIdx.x;
  const int b   = bh / NH_DIM;
  const int h   = bh % NH_DIM;
  const int kvh = h / QPK;
  const int q0  = blockIdx.y * 64;

  // Q fragments for this wave's 16 rows, 4 d-steps of 32 (reused for all KV tiles)
  Frag16 qf[4];
  {
    int qrow = q0 + wave * 16 + l16;
    const unsigned short* qp =
        &Qb[(((size_t)b * L_DIM + qrow) * NH_DIM + h) * HD_DIM];
    int kh = lhalf * 8;
#pragma unroll
    for (int s = 0; s < 4; ++s) {
      qf[s].u[0] = *reinterpret_cast<const uint4*>(qp + s * 32 + kh);
      qf[s].u[1] = *reinterpret_cast<const uint4*>(qp + s * 32 + kh + 16);
    }
  }

  const int skey  = tid >> 2;                   // 0..31
  const int sdoff = (tid & 3) * 32;             // 0..96
  const size_t kvrow = ((size_t)b * NKV_DIM + kvh) * HD_DIM;
  auto stage = [&](int kt, int buf) {           // 8 async b128 per thread
    const int kbase = kt * 32;
    const unsigned short* kg =
        &Kb[(((size_t)b * L_DIM + kbase + skey) * NKV_DIM + kvh) * HD_DIM + sdoff];
    unsigned short* kl = &Ks[buf][skey][sdoff];
#pragma unroll
    for (int c = 0; c < 4; ++c) async_copy16(kg + c * 8, kl + c * 8);
    const unsigned short* vg = &Vtg[(kvrow + tid) * L_DIM + kbase];
    unsigned short* vl = &Vt[buf][tid][0];
#pragma unroll
    for (int c = 0; c < 4; ++c) async_copy16(vg + c * 8, vl + c * 8);
  };

  v8f o[8];
#pragma unroll
  for (int i = 0; i < 8; ++i) o[i] = zero8();
  float mrow[8], lrow[8];
#pragma unroll
  for (int v = 0; v < 8; ++v) { mrow[v] = -1e30f; lrow[v] = 0.0f; }

  const int ntiles = q0 / 32 + 2;               // causal: keys <= q0+63
  stage(0, 0);
  for (int kt = 0; kt < ntiles; ++kt) {
    const int buf = kt & 1;
    const int kbase = kt * 32;
    __syncthreads();                            // buf^1 readers done
    if (kt + 1 < ntiles) {
      stage(kt + 1, buf ^ 1);
      ASYNC_WAIT(8);                            // in-order: tile kt complete
    } else {
      ASYNC_WAIT(0);
    }
    __syncthreads();

    // S = Q @ K^T  (2 key n-tiles x 4 d k-steps)
    v8f s[2];
    s[0] = zero8(); s[1] = zero8();
#pragma unroll
    for (int j = 0; j < 2; ++j) {
      int key = j * 16 + l16;
      int kst = lhalf * 8;
#pragma unroll
      for (int ds = 0; ds < 4; ++ds) {
        Frag16 kf;
        kf.u[0] = *reinterpret_cast<const uint4*>(&Ks[buf][key][ds * 32 + kst]);
        kf.u[1] = *reinterpret_cast<const uint4*>(&Ks[buf][key][ds * 32 + kst + 16]);
        s[j] = __builtin_amdgcn_wmma_f32_16x16x32_bf16(
            false, qf[ds].v, false, kf.v, (short)0, s[j], false, false);
      }
    }

    // scale + causal/key mask
    float sc[2][8];
#pragma unroll
    for (int j = 0; j < 2; ++j) {
      int kglob = kbase + j * 16 + l16;
      int msk = amask[(size_t)b * L_DIM + kglob];
#pragma unroll
      for (int v = 0; v < 8; ++v) {
        int qglob = q0 + wave * 16 + lhalf * 8 + v;
        float val = s[j][v] * ATT_SCALE;
        if (kglob > qglob || msk == 0) val = -1e30f;
        sc[j][v] = val;
      }
    }

    // online softmax (rows live across 16-lane groups)
    float alpha[8];
#pragma unroll
    for (int v = 0; v < 8; ++v) {
      float mx = fmaxf(sc[0][v], sc[1][v]);
#pragma unroll
      for (int m = 1; m < 16; m <<= 1) mx = fmaxf(mx, __shfl_xor(mx, m, 16));
      float mnew = fmaxf(mrow[v], mx);
      alpha[v] = __expf(mrow[v] - mnew);
      mrow[v] = mnew;
      float p0 = __expf(sc[0][v] - mnew);
      float p1 = __expf(sc[1][v] - mnew);
      int prow = wave * 16 + lhalf * 8 + v;
      Ps[prow][l16]      = f2bf(p0);
      Ps[prow][16 + l16] = f2bf(p1);
      float rs = p0 + p1;
#pragma unroll
      for (int m = 1; m < 16; m <<= 1) rs += __shfl_xor(rs, m, 16);
      lrow[v] = alpha[v] * lrow[v] + rs;
    }
#pragma unroll
    for (int nt = 0; nt < 8; ++nt)
#pragma unroll
      for (int v = 0; v < 8; ++v) o[nt][v] *= alpha[v];

    // P fragment back from LDS, then O += P @ V
    Frag16 pf;
    {
      int prow = wave * 16 + l16;
      int kh = lhalf * 8;
      pf.u[0] = *reinterpret_cast<const uint4*>(&Ps[prow][kh]);
      pf.u[1] = *reinterpret_cast<const uint4*>(&Ps[prow][kh + 16]);
    }
#pragma unroll
    for (int nt = 0; nt < 8; ++nt) {
      Frag16 vf;
      int dcol = nt * 16 + l16;
      int kst = lhalf * 8;
      vf.u[0] = *reinterpret_cast<const uint4*>(&Vt[buf][dcol][kst]);
      vf.u[1] = *reinterpret_cast<const uint4*>(&Vt[buf][dcol][kst + 16]);
      o[nt] = __builtin_amdgcn_wmma_f32_16x16x32_bf16(
          false, pf.v, false, vf.v, (short)0, o[nt], false, false);
    }
  }

  // normalize + store
#pragma unroll
  for (int v = 0; v < 8; ++v) {
    float inv = lrow[v] > 0.0f ? 1.0f / lrow[v] : 0.0f;
    int qrow = q0 + wave * 16 + lhalf * 8 + v;
    float* yp = &Y[(((size_t)b * L_DIM + qrow) * NH_DIM + h) * HD_DIM];
#pragma unroll
    for (int nt = 0; nt < 8; ++nt) yp[nt * 16 + l16] = o[nt][v] * inv;
  }
}

// ---------------- host side ----------------
static void launch_cvt(const float* in, unsigned short* out, size_t n,
                       hipStream_t s) {
  size_t n4 = n / 4;
  int blocks = (int)((n4 + 255) / 256);
  cvt_f32_bf16_x4<<<blocks, 256, 0, s>>>(in, out, (unsigned long long)n4);
}

static void launch_transpose(const float* in, unsigned short* out, int K, int N,
                             hipStream_t s) {
  dim3 grid(N / 32, K / 32);
  transpose_f32_bf16<<<grid, 256, 0, s>>>(in, out, K, N);
}

static void launch_gemm(const unsigned short* A, const unsigned short* Bt,
                        float* C, int M, int N, int K, hipStream_t s) {
  dim3 grid(N / BN, M / BM);
  gemm_bf16_wmma<<<grid, 256, 0, s>>>(A, Bt, C, M, N, K);
}

extern "C" void kernel_launch(void* const* d_in, const int* in_sizes, int n_in,
                              void* d_out, int out_size, void* d_ws,
                              size_t ws_size, hipStream_t stream) {
  const float* x  = (const float*)d_in[0];
  const float* Wq = (const float*)d_in[1];
  const float* Wc = (const float*)d_in[2];
  const float* Wk = (const float*)d_in[3];
  const float* Wv = (const float*)d_in[4];
  const float* Wo = (const float*)d_in[5];
  const int* amask = (const int*)d_in[6];
  float* out = (float*)d_out;

  const size_t M  = (size_t)B_DIM * L_DIM;          // 4096
  const size_t nx = M * HID_DIM;
  const size_t nWq = (size_t)HID_DIM * NH_DIM * HD_DIM;
  const size_t nWc = (size_t)HID_DIM * LAT_DIM;
  const size_t nWk = (size_t)LAT_DIM * NKV_DIM * HD_DIM;
  const size_t nWv = nWk;
  const size_t nWo = (size_t)NH_DIM * HD_DIM * HID_DIM;
  const size_t nq = M * NH_DIM * HD_DIM;
  const size_t nc = M * LAT_DIM;
  const size_t nk = M * NKV_DIM * HD_DIM;
  const size_t nv = nk;
  const size_t ny = nq;

  char* w = (char*)d_ws;
  auto carve = [&](size_t bytes) {
    char* p = w;
    w += (bytes + 255) & ~(size_t)255;
    return p;
  };
  unsigned short* xb  = (unsigned short*)carve(nx * 2);
  unsigned short* Wqt = (unsigned short*)carve(nWq * 2);   // [N][K] bf16
  unsigned short* Wct = (unsigned short*)carve(nWc * 2);
  unsigned short* Wkt = (unsigned short*)carve(nWk * 2);
  unsigned short* Wvt = (unsigned short*)carve(nWv * 2);
  unsigned short* Wot = (unsigned short*)carve(nWo * 2);
  float* qf = (float*)carve(nq * 4);
  float* cf = (float*)carve(nc * 4);
  unsigned short* cb = (unsigned short*)carve(nc * 2);
  float* kf = (float*)carve(nk * 4);
  float* vf = (float*)carve(nv * 4);
  unsigned short* qb = (unsigned short*)carve(nq * 2);
  unsigned short* kb = (unsigned short*)carve(nk * 2);
  unsigned short* vt = (unsigned short*)carve(nv * 2);     // [B,NKV,HD,L] bf16
  float* yf = (float*)carve(ny * 4);
  unsigned short* yb = (unsigned short*)carve(ny * 2);

  // 1. bf16 activation copy + transposed bf16 weights
  launch_cvt(x, xb, nx, stream);
  launch_transpose(Wq, Wqt, HID_DIM, NH_DIM * HD_DIM, stream);
  launch_transpose(Wc, Wct, HID_DIM, LAT_DIM, stream);
  launch_transpose(Wk, Wkt, LAT_DIM, NKV_DIM * HD_DIM, stream);
  launch_transpose(Wv, Wvt, LAT_DIM, NKV_DIM * HD_DIM, stream);
  launch_transpose(Wo, Wot, NH_DIM * HD_DIM, HID_DIM, stream);

  // 2. projections
  launch_gemm(xb, Wqt, qf, (int)M, NH_DIM * HD_DIM, HID_DIM, stream); // q
  launch_gemm(xb, Wct, cf, (int)M, LAT_DIM, HID_DIM, stream);         // c
  launch_cvt(cf, cb, nc, stream);
  launch_gemm(cb, Wkt, kf, (int)M, NKV_DIM * HD_DIM, LAT_DIM, stream); // k
  launch_gemm(cb, Wvt, vf, (int)M, NKV_DIM * HD_DIM, LAT_DIM, stream); // v

  // 3. RoPE (fused bf16 cast) + V transpose
  {
    unsigned long long tq = (unsigned long long)B_DIM * L_DIM * NH_DIM * 64;
    rope_to_bf16<<<(unsigned)(tq / 256), 256, 0, stream>>>(qf, qb, NH_DIM);
    unsigned long long tk = (unsigned long long)B_DIM * L_DIM * NKV_DIM * 64;
    rope_to_bf16<<<(unsigned)(tk / 256), 256, 0, stream>>>(kf, kb, NKV_DIM);
  }
  {
    dim3 grid(HD_DIM / 32, L_DIM / 32, B_DIM * NKV_DIM);
    transpose_v_bf16<<<grid, 256, 0, stream>>>(vf, vt);
  }

  // 4. flash attention
  {
    dim3 grid(B_DIM * NH_DIM, L_DIM / 64);
    flash_attn_wmma<<<grid, 128, 0, stream>>>(qb, kb, vt, amask, yf);
  }

  // 5. output projection -> d_out
  launch_cvt(yf, yb, ny, stream);
  launch_gemm(yb, Wot, out, (int)M, HID_DIM, NH_DIM * HD_DIM, stream);
}
